// SequentialBlock_64802466562167
// MI455X (gfx1250) — compile-verified, matrix operand order checked by reading, not run on hardware
//
#include <hip/hip_runtime.h>
#include <math.h>

#define B_    2
#define L_    2048
#define D_    768
#define K_    24
#define KU_   3
#define KY_   2
#define HDIM_ 2048

typedef __bf16 bf16;
typedef __attribute__((ext_vector_type(8)))  __bf16 v8bf;
typedef __attribute__((ext_vector_type(16))) __bf16 v16bf;
typedef __attribute__((ext_vector_type(8)))  float  v8f;
typedef __attribute__((ext_vector_type(4)))  unsigned int v4u;
typedef __attribute__((ext_vector_type(8)))  int v8i;
typedef __attribute__((ext_vector_type(4)))  int v4i;

#ifndef __has_builtin
#define __has_builtin(x) 0
#endif
#if __has_builtin(__builtin_amdgcn_tensor_load_to_lds) && \
    __has_builtin(__builtin_amdgcn_s_wait_tensorcnt)
#define HAVE_TDM 1
#else
#define HAVE_TDM 0
#endif

#if HAVE_TDM
// Low 32 bits of a generic LDS address are the workgroup-relative LDS offset
// (flat aperture check uses addr[63:32] only; see ISA 10.2).
__device__ __forceinline__ unsigned lds_off(const void* p) {
    return (unsigned)(unsigned long long)p;
}

// 2D tile DMA: global (row-major, stride0 in elements of 2 bytes) -> LDS
// (contiguous tile0 x tile1). D# packed per ISA 8.3/8.4.
__device__ __forceinline__ void tdm_load_2d(unsigned ldsOff, const void* gp,
                                            unsigned tile0, unsigned tile1,
                                            unsigned dim0, unsigned dim1,
                                            unsigned long long stride0)
{
    unsigned long long ga = (unsigned long long)gp;
    v4u g0;
    g0[0] = 1u;                                   // count=1 (valid user D#)
    g0[1] = ldsOff;                               // lds_addr
    g0[2] = (unsigned)(ga & 0xffffffffu);         // global_addr[31:0]
    g0[3] = (unsigned)((ga >> 32) & 0x01ffffffu)  // global_addr[56:32]
          | (2u << 30);                           // type=2 ("image")
    v8i g1;
    g1[0] = (int)(1u << 16);                      // data_size=1 -> 2 bytes
    g1[1] = (int)((dim0 & 0xffffu) << 16);        // tensor_dim0[15:0]
    g1[2] = (int)((dim0 >> 16) | ((dim1 & 0xffffu) << 16));
    g1[3] = (int)((dim1 >> 16) | (tile0 << 16));  // tile_dim0
    g1[4] = (int)(tile1 & 0xffffu);               // tile_dim1 (tile_dim2=0)
    g1[5] = (int)(unsigned)(stride0 & 0xffffffffu);
    g1[6] = (int)(unsigned)((stride0 >> 32) & 0xffffu);
    g1[7] = 0;
    v4i z4 = {0, 0, 0, 0};
#if defined(__clang_major__) && (__clang_major__ >= 23)
    v8i z8 = {0, 0, 0, 0, 0, 0, 0, 0};
    __builtin_amdgcn_tensor_load_to_lds(g0, g1, z4, z4, z8, 0);
#else
    __builtin_amdgcn_tensor_load_to_lds(g0, g1, z4, z4, 0);
#endif
}
#endif // HAVE_TDM

// -------- WMMA fragment loader (ISA 7.12.2, 16-bit A/B 16x32 layout) -------
// lanes 0-15 : K = 0..7 , 16..23   (row = lane&15)
// lanes 16-31: K = 8..15, 24..31
__device__ __forceinline__ v16bf frag_ld(const bf16* rowp, bool hi) {
    const bf16* p = rowp + (hi ? 8 : 0);
    v8bf a = *(const v8bf*)p;
    v8bf b = *(const v8bf*)(p + 16);
    v16bf r;
#pragma unroll
    for (int i = 0; i < 8; ++i) { r[i] = a[i]; r[i + 8] = b[i]; }
    return r;
}

// ---------------------------------------------------------------------------
// Generic bf16 GEMM: C[M,N](f32) (+=) A[M,Kd](bf16,lda) @ B[Kd,N](bf16,ldb)
// Block 256 thr = 8 waves; tile 128(M) x 128(N), K-step 32.
// Wave w: m-pair = w&3 (2 x 16 rows), n-group = w>>2 (4 x 16 cols) -> 8 WMMA.
// A tile staged by the Tensor Data Mover when shift==0; manual otherwise.
// ---------------------------------------------------------------------------
__global__ __launch_bounds__(256)
void k_gemm(const bf16* __restrict__ A, int lda,
            const bf16* __restrict__ Bm, int ldb,
            float* __restrict__ C, int ldc,
            const float* __restrict__ addsrc,
            int M, int N, int Kd, int shift, int seqL, int accum)
{
    __shared__ __align__(16) bf16 As[128][32];
    __shared__ __align__(16) bf16 Bs[128][40];   // [n][k], padded stride

    const int tid   = threadIdx.x;
    const int lane  = tid & 31;
    const int wv    = tid >> 5;
    const int mpair = wv & 3;
    const int ngrp  = wv >> 2;
    const int m0    = blockIdx.y * 128;
    const int n0    = blockIdx.x * 128;

    v8f acc[2][4];
#pragma unroll
    for (int mi = 0; mi < 2; ++mi)
#pragma unroll
        for (int j = 0; j < 4; ++j)
#pragma unroll
            for (int e = 0; e < 8; ++e) acc[mi][j][e] = 0.f;

    const int ar  = tid >> 1;          // A row 0..127
    const int ac0 = (tid & 1) * 16;    // A col start (16 elems = 32B)
    const int bk  = tid & 31;          // B k 0..31
    const int bn0 = (tid >> 5) * 16;   // B n chunk
    const bool hi = (lane >= 16);
    const int  rr = lane & 15;

#if HAVE_TDM
    const bool tdmA = (shift == 0);
    const unsigned asOff = lds_off(&As[0][0]);
#else
    const bool tdmA = false;
#endif

    for (int k0 = 0; k0 < Kd; k0 += 32) {
        __syncthreads();
        if (tdmA) {
#if HAVE_TDM
            if (wv == 0)
                tdm_load_2d(asOff, A + (size_t)m0 * lda + k0, 32, 128,
                            (unsigned)Kd, (unsigned)M,
                            (unsigned long long)(unsigned)lda);
#endif
        } else {
            int gr = m0 + ar;
            if ((gr % seqL) >= shift) {
                const bf16* gp = A + (size_t)(gr - shift) * lda + k0 + ac0;
                *(v8bf*)(&As[ar][ac0])     = *(const v8bf*)gp;
                *(v8bf*)(&As[ar][ac0 + 8]) = *(const v8bf*)(gp + 8);
            } else {
#pragma unroll
                for (int e = 0; e < 16; ++e) As[ar][ac0 + e] = (bf16)0.f;
            }
        }
        { // stage B tile transposed -> [n][k]
            const bf16* gp = Bm + (size_t)(k0 + bk) * ldb + n0 + bn0;
            v8bf x0 = *(const v8bf*)gp;
            v8bf x1 = *(const v8bf*)(gp + 8);
#pragma unroll
            for (int e = 0; e < 8; ++e) {
                Bs[bn0 + e][bk]     = x0[e];
                Bs[bn0 + 8 + e][bk] = x1[e];
            }
        }
        if (k0 + 32 < Kd)
            __builtin_prefetch(Bm + (size_t)(k0 + 32 + bk) * ldb + n0 + bn0, 0, 1);
#if HAVE_TDM
        if (tdmA && wv == 0) __builtin_amdgcn_s_wait_tensorcnt(0);
#endif
        __syncthreads();

        v16bf af0 = frag_ld(&As[mpair * 32 + rr][0], hi);
        v16bf af1 = frag_ld(&As[mpair * 32 + 16 + rr][0], hi);
#pragma unroll
        for (int j = 0; j < 4; ++j) {
            v16bf bf = frag_ld(&Bs[ngrp * 64 + j * 16 + rr][0], hi);
            acc[0][j] = __builtin_amdgcn_wmma_f32_16x16x32_bf16(
                false, af0, false, bf, (short)0, acc[0][j], false, false);
            acc[1][j] = __builtin_amdgcn_wmma_f32_16x16x32_bf16(
                false, af1, false, bf, (short)0, acc[1][j], false, false);
        }
    }

    // C/D layout: lanes 0-15 N=lane, M=r ; lanes 16-31 N=lane-16, M=8+r
#pragma unroll
    for (int mi = 0; mi < 2; ++mi)
#pragma unroll
        for (int j = 0; j < 4; ++j) {
            int ncol = n0 + ngrp * 64 + j * 16 + rr;
#pragma unroll
            for (int r = 0; r < 8; ++r) {
                int    mrow = m0 + mpair * 32 + mi * 16 + r + (hi ? 8 : 0);
                size_t idx  = (size_t)mrow * ldc + ncol;
                float  v    = acc[mi][j][r];
                if (addsrc) v += addsrc[idx];
                if (accum)  v += C[idx];
                C[idx] = v;
            }
        }
}

// ---------------------------------------------------------------------------
// Spectral conv as Toeplitz GEMM: Xt[b,l,k,d] = sum_s v'[l-s]*h[b,s,d].
// Reversed filter rvp makes the Toeplitz tile a TDM 2D tile with dim0
// stride 1 (LDS row i = Toeplitz row 127-i); frag loader flips the row.
// ---------------------------------------------------------------------------
__global__ __launch_bounds__(256)
void k_conv(const bf16* __restrict__ rvp, const bf16* __restrict__ h,
            bf16* __restrict__ xt)
{
    __shared__ __align__(16) bf16 As[128][32];
    __shared__ __align__(16) bf16 Hs[128][40];

    const int tid   = threadIdx.x;
    const int lane  = tid & 31;
    const int wv    = tid >> 5;
    const int mpair = wv & 3;
    const int ngrp  = wv >> 2;
    const int d0    = blockIdx.x * 128;
    const int l0    = blockIdx.y * 128;
    const int b     = blockIdx.z / K_;
    const int kf    = blockIdx.z % K_;

    const bf16* vk = rvp + (size_t)kf * (2 * L_);
    const bf16* hb = h + (size_t)b * L_ * D_;

    v8f acc[2][4];
#pragma unroll
    for (int mi = 0; mi < 2; ++mi)
#pragma unroll
        for (int j = 0; j < 4; ++j)
#pragma unroll
            for (int e = 0; e < 8; ++e) acc[mi][j][e] = 0.f;

    const int bk  = tid & 31;
    const int bn0 = (tid >> 5) * 16;
    const bool hi = (lane >= 16);
    const int  rr = lane & 15;

#if HAVE_TDM
    const unsigned asOff = lds_off(&As[0][0]);
#endif

    const int nsteps = l0 / 32 + 4;     // s-tiles up to the causal diagonal
    for (int st = 0; st < nsteps; ++st) {
        const int s0 = st * 32;
        const int t0 = L_ - 128 + s0 - l0;   // window base into rvp, >= 0
        __syncthreads();
#if HAVE_TDM
        if (wv == 0)
            tdm_load_2d(asOff, vk + t0, 32, 128,
                        2u * L_, 2u * L_, 1ull);
#else
        {
            int i = tid >> 1, c0 = (tid & 1) * 16;
#pragma unroll
            for (int e = 0; e < 16; ++e) As[i][c0 + e] = vk[t0 + i + c0 + e];
        }
#endif
        { // H tile transposed -> Hs[d][s]
            const bf16* gp = hb + (size_t)(s0 + bk) * D_ + d0 + bn0;
            v8bf x0 = *(const v8bf*)gp;
            v8bf x1 = *(const v8bf*)(gp + 8);
#pragma unroll
            for (int e = 0; e < 8; ++e) {
                Hs[bn0 + e][bk]     = x0[e];
                Hs[bn0 + 8 + e][bk] = x1[e];
            }
        }
#if HAVE_TDM
        if (wv == 0) __builtin_amdgcn_s_wait_tensorcnt(0);
#endif
        __syncthreads();

        // Toeplitz row r is at LDS row 127-r
        v16bf af0 = frag_ld(&As[127 - (mpair * 32 + rr)][0], hi);
        v16bf af1 = frag_ld(&As[127 - (mpair * 32 + 16 + rr)][0], hi);
#pragma unroll
        for (int j = 0; j < 4; ++j) {
            v16bf bf = frag_ld(&Hs[ngrp * 64 + j * 16 + rr][0], hi);
            acc[0][j] = __builtin_amdgcn_wmma_f32_16x16x32_bf16(
                false, af0, false, bf, (short)0, acc[0][j], false, false);
            acc[1][j] = __builtin_amdgcn_wmma_f32_16x16x32_bf16(
                false, af1, false, bf, (short)0, acc[1][j], false, false);
        }
    }

#pragma unroll
    for (int mi = 0; mi < 2; ++mi)
#pragma unroll
        for (int j = 0; j < 4; ++j) {
            int d = d0 + ngrp * 64 + j * 16 + rr;
#pragma unroll
            for (int r = 0; r < 8; ++r) {
                int l = l0 + mpair * 32 + mi * 16 + r + (hi ? 8 : 0);
                xt[(((size_t)(b * L_ + l)) * K_ + kf) * D_ + d] =
                    (bf16)acc[mi][j][r];
            }
        }
}

// ------------------------------ LayerNorm ----------------------------------
__global__ __launch_bounds__(256)
void k_ln(const float* __restrict__ x, const float* __restrict__ w,
          bf16* __restrict__ h)
{
    const int row = blockIdx.x;
    const float* xr = x + (size_t)row * D_;
    float s = 0.f, s2 = 0.f;
    for (int i = threadIdx.x; i < D_; i += 256) {
        float v = xr[i]; s += v; s2 += v * v;
    }
    __shared__ float sh1[256], sh2[256];
    sh1[threadIdx.x] = s; sh2[threadIdx.x] = s2;
    __syncthreads();
    for (int o = 128; o > 0; o >>= 1) {
        if (threadIdx.x < o) {
            sh1[threadIdx.x] += sh1[threadIdx.x + o];
            sh2[threadIdx.x] += sh2[threadIdx.x + o];
        }
        __syncthreads();
    }
    float mu  = sh1[0] / D_;
    float var = sh2[0] / D_ - mu * mu;
    float inv = rsqrtf(var + 1e-5f);
    for (int i = threadIdx.x; i < D_; i += 256)
        h[(size_t)row * D_ + i] = (bf16)((xr[i] - mu) * inv * w[i]);
}

// -------- reversed filter: rvp[k][t] = (t<L) ? evec[L-1-t,k]*ev[k]^.25 : 0 -
__global__ void k_vpad(const float* __restrict__ evec, const float* __restrict__ ev,
                       bf16* __restrict__ rvp)
{
    int idx = blockIdx.x * 256 + threadIdx.x;
    if (idx >= K_ * 2 * L_) return;
    int k = idx / (2 * L_), t = idx % (2 * L_);
    float v = 0.f;
    if (t < L_) v = evec[(size_t)(L_ - 1 - t) * K_ + k] * sqrtf(sqrtf(ev[k]));
    rvp[idx] = (bf16)v;
}

// ------------------- weight conversions ------------------------------------
__global__ void k_cvt(const float* __restrict__ s, bf16* __restrict__ d, int n)
{
    int i = blockIdx.x * 256 + threadIdx.x;
    if (i < n) d[i] = (bf16)s[i];
}

// mu_bf[j][i][o] = m_u[o][i][j]  (B-matrix layout for the AR GEMMs)
__global__ void k_mu_pack(const float* __restrict__ mu, bf16* __restrict__ d)
{
    int idx = blockIdx.x * 256 + threadIdx.x;
    if (idx >= KU_ * D_ * D_) return;
    int j = idx / (D_ * D_), r = idx % (D_ * D_);
    int i = r / D_, o = r % D_;
    d[idx] = (bf16)mu[((size_t)o * D_ + i) * KU_ + j];
}

// -------------- serial recurrence y_t = M0 y_{t-1} + M1 y_{t-2} + d_t ------
__global__ __launch_bounds__(768)
void k_recur(const float* __restrict__ delta, const float* __restrict__ my,
             float* __restrict__ y, bf16* __restrict__ ybf)
{
    const int b = blockIdx.x;
    const int o = threadIdx.x;
    __shared__ float y1[D_], y2[D_];
    y1[o] = 0.f; y2[o] = 0.f;
    __syncthreads();
    const float* r0 = my + ((size_t)o * KY_ + 0) * D_;
    const float* r1 = my + ((size_t)o * KY_ + 1) * D_;
    for (int t = 0; t < L_; ++t) {
        float acc = delta[((size_t)b * L_ + t) * D_ + o];
#pragma unroll 4
        for (int i = 0; i < D_; ++i) acc += r0[i] * y1[i] + r1[i] * y2[i];
        __syncthreads();
        y2[o] = y1[o];
        y1[o] = acc;
        size_t idx = ((size_t)b * L_ + t) * D_ + o;
        y[idx]   = acc;
        ybf[idx] = (bf16)acc;
        __syncthreads();
    }
}

// ------------------------- SwiGLU gate -------------------------------------
__global__ void k_silu(const float* __restrict__ t1, const float* __restrict__ t2,
                       bf16* __restrict__ s, int n)
{
    int i = blockIdx.x * 256 + threadIdx.x;
    if (i < n) {
        float a = t1[i];
        s[i] = (bf16)((a / (1.f + __expf(-a))) * t2[i]);
    }
}

// ---------------------------------------------------------------------------
extern "C" void kernel_launch(void* const* d_in, const int* in_sizes, int n_in,
                              void* d_out, int out_size, void* d_ws, size_t ws_size,
                              hipStream_t stream)
{
    (void)in_sizes; (void)n_in; (void)out_size; (void)ws_size;

    const float* x     = (const float*)d_in[0];
    const float* ln_w  = (const float*)d_in[1];
    const float* m_u   = (const float*)d_in[2];
    const float* m_phi = (const float*)d_in[3];
    const float* m_y   = (const float*)d_in[4];
    const float* w1    = (const float*)d_in[5];
    const float* v1    = (const float*)d_in[6];
    const float* w2    = (const float*)d_in[7];
    const float* ev    = (const float*)d_in[8];
    const float* evec  = (const float*)d_in[9];
    float*       out   = (float*)d_out;

    char*  ws  = (char*)d_ws;
    size_t off = 0;
    auto alloc = [&](size_t bytes) -> char* {
        char* p = ws + off;
        off = (off + bytes + 255) & ~(size_t)255;
        return p;
    };

    const size_t NB = (size_t)B_ * L_;            // 4096 rows
    bf16*  rvp   = (bf16*)alloc((size_t)K_ * 2 * L_ * 2);
    bf16*  hbf   = (bf16*)alloc(NB * D_ * 2);
    bf16*  mphib = (bf16*)alloc((size_t)K_ * D_ * D_ * 2);
    bf16*  mub   = (bf16*)alloc((size_t)KU_ * D_ * D_ * 2);
    bf16*  w1b   = (bf16*)alloc((size_t)D_ * HDIM_ * 2);
    bf16*  v1b   = (bf16*)alloc((size_t)D_ * HDIM_ * 2);
    bf16*  w2b   = (bf16*)alloc((size_t)HDIM_ * D_ * 2);
    float* delta = (float*)alloc(NB * D_ * 4);
    float* yv    = (float*)alloc(NB * D_ * 4);
    bf16*  ybf   = (bf16*)alloc(NB * D_ * 2);
    // Xt (151 MB) and SwiGLU temporaries (84 MB) are disjoint in time:
    char*  big   = alloc(NB * (size_t)K_ * D_ * 2);
    bf16*  xt    = (bf16*)big;
    float* t1    = (float*)big;
    float* t2    = (float*)(big + NB * (size_t)HDIM_ * 4);
    bf16*  sbf   = (bf16*)(big + 2 * NB * (size_t)HDIM_ * 4);

    auto g1 = [](size_t n) { return dim3((unsigned)((n + 255) / 256)); };

    // weight prep (independent)
    k_cvt<<<g1((size_t)K_ * D_ * D_), 256, 0, stream>>>(m_phi, mphib, K_ * D_ * D_);
    k_cvt<<<g1((size_t)D_ * HDIM_), 256, 0, stream>>>(w1, w1b, D_ * HDIM_);
    k_cvt<<<g1((size_t)D_ * HDIM_), 256, 0, stream>>>(v1, v1b, D_ * HDIM_);
    k_cvt<<<g1((size_t)HDIM_ * D_), 256, 0, stream>>>(w2, w2b, HDIM_ * D_);
    k_mu_pack<<<g1((size_t)KU_ * D_ * D_), 256, 0, stream>>>(m_u, mub);
    k_vpad<<<g1((size_t)K_ * 2 * L_), 256, 0, stream>>>(evec, ev, rvp);
    k_ln<<<(unsigned)NB, 256, 0, stream>>>(x, ln_w, hbf);

    // spectral conv: Xt[b,l,k,d]
    k_conv<<<dim3(D_ / 128, L_ / 128, B_ * K_), 256, 0, stream>>>(rvp, hbf, xt);

    // delta = Xt_flat[NB, K*D] @ m_phi[K*D, D]
    k_gemm<<<dim3(D_ / 128, NB / 128), 256, 0, stream>>>(
        xt, K_ * D_, mphib, D_, delta, D_, nullptr,
        (int)NB, D_, K_ * D_, 0, 1, 0);

    // delta += sum_j shift_j(h) @ m_u_j^T   (causal AR term)
    for (int j = 0; j < KU_; ++j)
        k_gemm<<<dim3(D_ / 128, NB / 128), 256, 0, stream>>>(
            hbf, D_, mub + (size_t)j * D_ * D_, D_, delta, D_, nullptr,
            (int)NB, D_, D_, j, L_, 1);

    // serial recurrence
    k_recur<<<B_, D_, 0, stream>>>(delta, m_y, yv, ybf);

    // SwiGLU
    k_gemm<<<dim3(HDIM_ / 128, NB / 128), 256, 0, stream>>>(
        ybf, D_, w1b, HDIM_, t1, HDIM_, nullptr, (int)NB, HDIM_, D_, 0, 1, 0);
    k_gemm<<<dim3(HDIM_ / 128, NB / 128), 256, 0, stream>>>(
        ybf, D_, v1b, HDIM_, t2, HDIM_, nullptr, (int)NB, HDIM_, D_, 0, 1, 0);
    k_silu<<<g1(NB * HDIM_), 256, 0, stream>>>(t1, t2, sbf, (int)(NB * HDIM_));

    // out = x + s @ w2
    k_gemm<<<dim3(D_ / 128, NB / 128), 256, 0, stream>>>(
        sbf, HDIM_, w2b, D_, out, D_, x, (int)NB, D_, HDIM_, 0, 1, 0);
}
